// DifferentiableDAG_90237262889741
// MI455X (gfx1250) — compile-verified, matrix operand order checked by reading, not run on hardware
//
#include <hip/hip_runtime.h>
#include <math.h>

// ---------------------------------------------------------------------------
// Types for CDNA5 WMMA
// ---------------------------------------------------------------------------
typedef __attribute__((ext_vector_type(16))) __bf16 v16bf;
typedef __attribute__((ext_vector_type(8)))  float  v8f;

static __device__ __forceinline__ __bf16 f2bf(float x) { return (__bf16)x; }

#define BB     8192
#define HH     512
#define NMAX   16
#define NOPS   9
#define OUTLD  513   // H + 1 (value appended)

// ---------------------------------------------------------------------------
// CDNA5 async global->LDS copy (ASYNCcnt-tracked, no VGPR staging).
// GV addressing: vdst = 32-bit LDS byte offset, vaddr = 64-bit global address.
// ---------------------------------------------------------------------------
static __device__ __forceinline__ void async_b128(const __bf16* g, __bf16* l)
{
    unsigned int loff      = (unsigned int)(unsigned long long)(void*)l;  // low 32b = LDS offset
    unsigned long long gaddr = (unsigned long long)(const void*)g;
    asm volatile("global_load_async_to_lds_b128 %0, %1, off"
                 :: "v"(loff), "v"(gaddr)
                 : "memory");
}
static __device__ __forceinline__ void wait_async0()
{
    asm volatile("s_wait_asynccnt 0x0" ::: "memory");
}

// ---------------------------------------------------------------------------
// WMMA bf16 GEMM:  C[M,N] (f32) = A[M,K](bf16,row) * Bt[N,K](bf16,row)^T
//                                 (+ bias[N]) (+= old C if accumulate)
// Tile: 256x64, K-step 32. 256 threads = 8 waves; each wave owns 32 rows x 64
// cols = 2 A-frags x 4 B-frags -> 8 back-to-back v_wmma per K-step.
// Double-buffered LDS fed by global_load_async_to_lds_b128; one barrier/K-step.
// Requires: M%256==0, N%64==0, K%32==0, 16B-aligned pointers.
// ---------------------------------------------------------------------------
#define TM 256
#define TN 64
#define TK 32
#define LDS_PAD 8
#define LDAS (TK + LDS_PAD)

__global__ __launch_bounds__(256) void k_gemm_bf16(
    const __bf16* __restrict__ A, int lda,
    const __bf16* __restrict__ Bt, int ldb,
    float* __restrict__ C, int ldc,
    int K,
    const float* __restrict__ bias,
    int accumulate)
{
    __shared__ __bf16 As[2][TM][LDAS];   // 2 x 256x40 bf16 = 40 KB
    __shared__ __bf16 Bs[2][TN][LDAS];   // 2 x  64x40 bf16 = 10 KB

    const int tid  = threadIdx.x;
    const int wave = tid >> 5;
    const int lane = tid & 31;
    const int m0   = blockIdx.x * TM;
    const int n0   = blockIdx.y * TN;

    // Per-thread async-copy assignments for one (A,B) tile pair:
    // A tile: 256x32 bf16 = 1024 x b128 -> 4 per thread; B tile: 256 x b128 -> 1.
    auto issue_tile = [&](int k0, __bf16 (*as)[LDAS], __bf16 (*bs)[LDAS]) {
#pragma unroll
        for (int i = 0; i < 4; ++i) {
            int idx = tid + i * 256;
            int r   = idx >> 2;
            int c8  = idx & 3;
            async_b128(A + (size_t)(m0 + r) * lda + k0 + c8 * 8, &as[r][c8 * 8]);
        }
        {
            int r  = tid >> 2;
            int c8 = tid & 3;
            async_b128(Bt + (size_t)(n0 + r) * ldb + k0 + c8 * 8, &bs[r][c8 * 8]);
        }
    };

    v8f acc[2][4];
#pragma unroll
    for (int a = 0; a < 2; ++a)
#pragma unroll
        for (int t = 0; t < 4; ++t)
#pragma unroll
            for (int r = 0; r < 8; ++r) acc[a][t][r] = 0.0f;

    const int ntiles = K / TK;
    issue_tile(0, As[0], Bs[0]);

    for (int it = 0; it < ntiles; ++it) {
        const int buf = it & 1;
        wait_async0();        // my wave's tile-`it` copies landed in LDS
        __syncthreads();      // everyone's landed; everyone done reading buf^1

        if (it + 1 < ntiles)
            issue_tile((it + 1) * TK, As[buf ^ 1], Bs[buf ^ 1]);  // overlap with WMMA

        // Preload ALL fragments, then issue the 8 WMMAs back-to-back.
        // A-frag ISA layout: lanes 0-15 hold K=kb..kb+7 and kb+16..kb+23.
        const int rsub = lane & 15;
        const int kb   = (lane >> 4) * 8;
        v16bf af[2];
#pragma unroll
        for (int a = 0; a < 2; ++a) {
            const int rowA = wave * 32 + a * 16 + rsub;
#pragma unroll
            for (int i = 0; i < 8; ++i) {
                af[a][i]     = As[buf][rowA][kb + i];
                af[a][i + 8] = As[buf][rowA][kb + 16 + i];
            }
        }
        v16bf bfr[4];
#pragma unroll
        for (int nt = 0; nt < 4; ++nt) {
            const int colB = nt * 16 + rsub;
#pragma unroll
            for (int i = 0; i < 8; ++i) {
                bfr[nt][i]     = Bs[buf][colB][kb + i];
                bfr[nt][i + 8] = Bs[buf][colB][kb + 16 + i];
            }
        }
#pragma unroll
        for (int a = 0; a < 2; ++a)
#pragma unroll
            for (int nt = 0; nt < 4; ++nt)
                acc[a][nt] = __builtin_amdgcn_wmma_f32_16x16x32_bf16(
                    false, af[a], false, bfr[nt], (short)0, acc[a][nt], false, false);
    }

    // Epilogue: VGPR r holds M=r (lanes 0-15) / M=8+r (lanes 16-31), N = lane&15.
    const int nin = lane & 15;
#pragma unroll
    for (int a = 0; a < 2; ++a) {
        const int mbase = m0 + wave * 32 + a * 16 + 8 * (lane >> 4);
#pragma unroll
        for (int nt = 0; nt < 4; ++nt) {
            const int n = n0 + nt * 16 + nin;
            const float bv = bias ? bias[n] : 0.0f;
#pragma unroll
            for (int r = 0; r < 8; ++r) {
                size_t ci = (size_t)(mbase + r) * ldc + n;
                float v = acc[a][nt][r] + bv;
                if (accumulate) v += C[ci];
                C[ci] = v;
            }
        }
    }
}

// ---------------------------------------------------------------------------
// Transpose + convert weight to bf16:  Wt[(rowOff+n)*ldt + k] = W[k*ldw + n]
// ---------------------------------------------------------------------------
__global__ void k_transpose_bf16(const float* __restrict__ W, int ldw,
                                 int K, int N,
                                 __bf16* __restrict__ Wt, int ldt, int rowOff)
{
    size_t t = (size_t)blockIdx.x * 256 + threadIdx.x;
    if (t >= (size_t)K * N) return;
    int k = (int)(t / N);
    int n = (int)(t % N);
    Wt[(size_t)(rowOff + n) * ldt + k] = f2bf(W[(size_t)k * ldw + n]);
}

// ---------------------------------------------------------------------------
// Fold rank-1 MLP inputs: u1 = s2e_w @ W1[1024:1536], u2 = s2e_w @ W1[1536:2048],
// c0 = s2e_b @ (W1c + W1d) + mlp_b1
// ---------------------------------------------------------------------------
__global__ void k_precompute(const float* __restrict__ s2e_w,
                             const float* __restrict__ s2e_b,
                             const float* __restrict__ mlp_w1,
                             const float* __restrict__ mlp_b1,
                             float* __restrict__ u1, float* __restrict__ u2,
                             float* __restrict__ c0)
{
    int h = blockIdx.x * 256 + threadIdx.x;
    if (h >= HH) return;
    float a = 0.f, b = 0.f, c = 0.f;
    for (int k = 0; k < HH; ++k) {
        float w1 = mlp_w1[(size_t)(2 * HH + k) * HH + h];
        float w2 = mlp_w1[(size_t)(3 * HH + k) * HH + h];
        a += s2e_w[k] * w1;
        b += s2e_w[k] * w2;
        c += s2e_b[k] * (w1 + w2);
    }
    u1[h] = a; u2[h] = b; c0[h] = c + mlp_b1[h];
}

// ---------------------------------------------------------------------------
// Scatter embeds0/values0 into d_out (stride 513) and bf16 embeds mirror.
// ---------------------------------------------------------------------------
__global__ void k_init(const float* __restrict__ embeds0,
                       const float* __restrict__ values0,
                       float* __restrict__ out,
                       __bf16* __restrict__ embbf)
{
    size_t t = (size_t)blockIdx.x * 256 + threadIdx.x;   // over B*8*512
    size_t bn = t >> 9; int h = (int)(t & 511);
    size_t b = bn >> 3; int n = (int)(bn & 7);
    float v = embeds0[t];
    size_t row = b * NMAX + n;
    out[row * OUTLD + h] = v;
    embbf[row * HH + h]  = f2bf(v);
    if (h == 0) out[row * OUTLD + HH] = values0[bn];
}

// ---------------------------------------------------------------------------
// summary = mean over valid rows, converted to bf16 for GEMM A operand.
// ---------------------------------------------------------------------------
__global__ void k_summary(const float* __restrict__ out,
                          __bf16* __restrict__ sumbf, int N, float inv)
{
    size_t t = (size_t)blockIdx.x * 256 + threadIdx.x;   // b*512 + h
    size_t b = t >> 9; int h = (int)(t & 511);
    float s = 0.f;
    for (int n = 0; n < N; ++n) s += out[(b * NMAX + n) * OUTLD + h];
    sumbf[t] = f2bf(s * inv);
}

// ---------------------------------------------------------------------------
// Fused attention + op selection + scalar ops + MLP-input assembly.
// One wave32 per batch row; 8 rows per 256-thread block.
// ---------------------------------------------------------------------------
__global__ __launch_bounds__(256) void k_attention(
    float* __restrict__ out,              // (B,16,513): embeds + value column
    const float* __restrict__ keys,       // (B,16,512)
    const float* __restrict__ q123,       // (B,1536) raw GEMM output
    const float* __restrict__ operand_ctx,
    const float* __restrict__ op_ctx,
    const float* __restrict__ emb_s,      // step_emb row (512)
    const float* __restrict__ qp1_b, const float* __restrict__ qp2_b,
    const float* __restrict__ opq_b,
    const float* __restrict__ opsel_w,    // (512,9)
    const float* __restrict__ opsel_b,    // (9)
    const float* __restrict__ u1, const float* __restrict__ u2,
    const float* __restrict__ c0,
    const float* __restrict__ mlp_w1,     // rows 2048..2056 used (op_w part)
    __bf16* __restrict__ e12bf,           // (B,1024)  [e1 | e2]
    float* __restrict__ esum,             // (B,512)   e1+e2
    float* __restrict__ mlp_in,           // (B,512)   preload rank-1 + op parts
    int N)
{
    const int lane = threadIdx.x & 31;
    const int wave = threadIdx.x >> 5;
    const size_t b = (size_t)blockIdx.x * 8 + wave;

    // queries with bias + context + step embedding; lane owns h = i*32 + lane
    float q1c[16], q2c[16], q3c[16];
#pragma unroll
    for (int i = 0; i < 16; ++i) {
        int h = i * 32 + lane;
        float oc = operand_ctx[b * HH + h];
        float es = emb_s[h];
        q1c[i] = q123[b * 1536 + h]           + qp1_b[h] + oc + es;
        q2c[i] = q123[b * 1536 + 512 + h]     + qp2_b[h] + oc + es;
        q3c[i] = q123[b * 1536 + 1024 + h]    + opq_b[h] + op_ctx[b * HH + h] + es;
    }

    // attention logits (3 queries x up to 16 keys)
    float l1[16], l2[16], l3[16];
#pragma unroll
    for (int n = 0; n < 16; ++n) {
        float s1 = 0.f, s2 = 0.f, s3 = 0.f;
        if (n < N) {
            const float* kr = keys + (b * NMAX + n) * HH;
#pragma unroll
            for (int i = 0; i < 16; ++i) {
                float kv = kr[i * 32 + lane];
                s1 += kv * q1c[i]; s2 += kv * q2c[i]; s3 += kv * q3c[i];
            }
        }
        l1[n] = s1; l2[n] = s2; l3[n] = s3;
    }
#pragma unroll
    for (int n = 0; n < 16; ++n) {
#pragma unroll
        for (int off = 16; off > 0; off >>= 1) {
            l1[n] += __shfl_xor(l1[n], off, 32);
            l2[n] += __shfl_xor(l2[n], off, 32);
            l3[n] += __shfl_xor(l3[n], off, 32);
        }
    }
    const float scale = 0.04419417382415922f;  // 1/sqrt(512)

    // softmax over n (replicated in every lane)
    float m1 = -3.4e38f, m2 = -3.4e38f, m3 = -3.4e38f;
#pragma unroll
    for (int n = 0; n < 16; ++n) {
        if (n < N) {
            m1 = fmaxf(m1, l1[n] * scale);
            m2 = fmaxf(m2, l2[n] * scale);
            m3 = fmaxf(m3, l3[n] * scale);
        }
    }
    float a1[16], a2[16], a3[16];
    float z1 = 0.f, z2 = 0.f, z3 = 0.f;
#pragma unroll
    for (int n = 0; n < 16; ++n) {
        if (n < N) {
            a1[n] = __expf(l1[n] * scale - m1);
            a2[n] = __expf(l2[n] * scale - m2);
            a3[n] = __expf(l3[n] * scale - m3);
        } else { a1[n] = 0.f; a2[n] = 0.f; a3[n] = 0.f; }
        z1 += a1[n]; z2 += a2[n]; z3 += a3[n];
    }
    float r1 = 1.f / z1, r2 = 1.f / z2, r3 = 1.f / z3;
#pragma unroll
    for (int n = 0; n < 16; ++n) { a1[n] *= r1; a2[n] *= r2; a3[n] *= r3; }

    // weighted sums over embeds + scalar values
    float e1c[16], e2c[16], occ[16];
#pragma unroll
    for (int i = 0; i < 16; ++i) { e1c[i] = 0.f; e2c[i] = 0.f; occ[i] = 0.f; }
    float v1 = 0.f, v2 = 0.f;
#pragma unroll
    for (int n = 0; n < 16; ++n) {
        if (n < N) {
            const float* er = out + (b * NMAX + n) * OUTLD;
            float vn = er[HH];
            v1 += a1[n] * vn; v2 += a2[n] * vn;
#pragma unroll
            for (int i = 0; i < 16; ++i) {
                float ev = er[i * 32 + lane];
                e1c[i] += a1[n] * ev; e2c[i] += a2[n] * ev; occ[i] += a3[n] * ev;
            }
        }
    }

    // op selection: (B,512)@(512,9) per row, wave-reduced
    float opw[NOPS];
    float mo = -3.4e38f;
#pragma unroll
    for (int j = 0; j < NOPS; ++j) {
        float s = 0.f;
#pragma unroll
        for (int i = 0; i < 16; ++i) {
            int h = i * 32 + lane;
            s += occ[i] * opsel_w[(size_t)h * NOPS + j];
        }
#pragma unroll
        for (int off = 16; off > 0; off >>= 1) s += __shfl_xor(s, off, 32);
        opw[j] = s + opsel_b[j];
        mo = fmaxf(mo, opw[j]);
    }
    float zo = 0.f;
#pragma unroll
    for (int j = 0; j < NOPS; ++j) { opw[j] = __expf(opw[j] - mo); zo += opw[j]; }
    float ro = 1.f / zo;
#pragma unroll
    for (int j = 0; j < NOPS; ++j) opw[j] *= ro;

    // 9 elementwise ops on (v1, v2)
    float yc = fminf(fmaxf(v2, -6.0f), 6.0f);
    float opsv[NOPS];
    opsv[0] = v1 + v2;
    opsv[1] = v1;
    opsv[2] = v1 * v2;
    opsv[3] = v1 - v2;
    opsv[4] = v1 / (v2 + 1e-8f);
    opsv[5] = powf(fabsf(v1) + 1e-6f, yc);
    opsv[6] = logf(fabsf(v1) + 1e-8f);
    opsv[7] = fmaxf(v1, v2);
    opsv[8] = fminf(v1, v2);
    float nv = 0.f;
#pragma unroll
    for (int j = 0; j < NOPS; ++j) nv += opw[j] * opsv[j];
    if (lane == 0) out[(b * NMAX + N) * OUTLD + HH] = nv;  // append value

    // outputs: e12 bf16, esum, and mlp_in preloaded with rank-1 + op_w parts
#pragma unroll
    for (int i = 0; i < 16; ++i) {
        int h = i * 32 + lane;
        float me = v1 * u1[h] + v2 * u2[h] + c0[h];
#pragma unroll
        for (int j = 0; j < NOPS; ++j)
            me += opw[j] * mlp_w1[(size_t)(4 * HH + j) * HH + h];
        mlp_in[b * HH + h] = me;
        e12bf[b * 1024 + h]       = f2bf(e1c[i]);
        e12bf[b * 1024 + 512 + h] = f2bf(e2c[i]);
        esum[b * HH + h] = e1c[i] + e2c[i];
    }
}

// ---------------------------------------------------------------------------
// Exact GELU + bf16 convert for the second MLP GEMM.
// ---------------------------------------------------------------------------
__global__ void k_gelu(const float* __restrict__ mlp_in, __bf16* __restrict__ hid)
{
    size_t t = (size_t)blockIdx.x * 256 + threadIdx.x;
    float x = mlp_in[t];
    float g = 0.5f * x * (1.0f + erff(x * 0.70710678118654752f));
    hid[t] = f2bf(g);
}

// ---------------------------------------------------------------------------
// new_emb = 0.5*(e1+e2) + delta; append to d_out row N and bf16 mirror.
// ---------------------------------------------------------------------------
__global__ void k_finalize(const float* __restrict__ esum,
                           const float* __restrict__ delta,
                           float* __restrict__ out,
                           __bf16* __restrict__ embbf, int N)
{
    size_t t = (size_t)blockIdx.x * 256 + threadIdx.x;   // b*512 + h
    size_t b = t >> 9; int h = (int)(t & 511);
    float ne = 0.5f * esum[t] + delta[t];
    out[(b * NMAX + N) * OUTLD + h] = ne;
    embbf[(b * NMAX + N) * HH + h]  = f2bf(ne);
}

// ---------------------------------------------------------------------------
// Host orchestration
// ---------------------------------------------------------------------------
extern "C" void kernel_launch(void* const* d_in, const int* in_sizes, int n_in,
                              void* d_out, int out_size, void* d_ws, size_t ws_size,
                              hipStream_t stream)
{
    const float* embeds0     = (const float*)d_in[0];
    const float* values0     = (const float*)d_in[1];
    const float* operand_ctx = (const float*)d_in[2];
    const float* op_ctx      = (const float*)d_in[3];
    const float* qp1_w = (const float*)d_in[4];  const float* qp1_b = (const float*)d_in[5];
    const float* qp2_w = (const float*)d_in[6];  const float* qp2_b = (const float*)d_in[7];
    const float* opq_w = (const float*)d_in[8];  const float* opq_b = (const float*)d_in[9];
    const float* key_w = (const float*)d_in[10]; const float* key_b = (const float*)d_in[11];
    const float* opsel_w = (const float*)d_in[12]; const float* opsel_b = (const float*)d_in[13];
    const float* step_emb = (const float*)d_in[14];
    const float* s2e_w = (const float*)d_in[15]; const float* s2e_b = (const float*)d_in[16];
    const float* mlp_w1 = (const float*)d_in[17]; const float* mlp_b1 = (const float*)d_in[18];
    const float* mlp_w2 = (const float*)d_in[19]; const float* mlp_b2 = (const float*)d_in[20];

    float* out = (float*)d_out;

    // Workspace carve-out
    char* wsb = (char*)d_ws;
    size_t off = 0;
    auto take = [&](size_t bytes) -> void* {
        void* p = wsb + off;
        off += (bytes + 255) & ~(size_t)255;
        return p;
    };
    __bf16* embbf  = (__bf16*)take((size_t)BB * NMAX * HH * 2);   // 128 MB
    float*  keys   = (float*) take((size_t)BB * NMAX * HH * 4);   // 256 MB
    __bf16* sumbf  = (__bf16*)take((size_t)BB * HH * 2);
    float*  q123   = (float*) take((size_t)BB * 3 * HH * 4);
    __bf16* e12bf  = (__bf16*)take((size_t)BB * 2 * HH * 2);
    float*  esum   = (float*) take((size_t)BB * HH * 4);
    float*  mlpin  = (float*) take((size_t)BB * HH * 4);
    __bf16* hidbf  = (__bf16*)take((size_t)BB * HH * 2);
    float*  delta  = (float*) take((size_t)BB * HH * 4);
    __bf16* keyWT  = (__bf16*)take((size_t)HH * HH * 2);
    __bf16* WqT    = (__bf16*)take((size_t)3 * HH * HH * 2);       // [qp1|qp2|opq]^T
    __bf16* W1abT  = (__bf16*)take((size_t)HH * 2 * HH * 2);       // mlp_w1[0:1024]^T
    __bf16* W2T    = (__bf16*)take((size_t)HH * HH * 2);
    float*  u1     = (float*) take((size_t)HH * 4);
    float*  u2     = (float*) take((size_t)HH * 4);
    float*  c0     = (float*) take((size_t)HH * 4);
    (void)ws_size; (void)in_sizes; (void)n_in; (void)out_size;

    // ---- one-time weight prep (cheap; deterministic, done every call) ----
    {
        int g = (HH * HH + 255) / 256;
        k_transpose_bf16<<<g, 256, 0, stream>>>(key_w, HH, HH, HH, keyWT, HH, 0);
        k_transpose_bf16<<<g, 256, 0, stream>>>(qp1_w, HH, HH, HH, WqT, HH, 0);
        k_transpose_bf16<<<g, 256, 0, stream>>>(qp2_w, HH, HH, HH, WqT, HH, 512);
        k_transpose_bf16<<<g, 256, 0, stream>>>(opq_w, HH, HH, HH, WqT, HH, 1024);
        k_transpose_bf16<<<(2 * HH * HH + 255) / 256, 256, 0, stream>>>(
            mlp_w1, HH, 2 * HH, HH, W1abT, 2 * HH, 0);
        k_transpose_bf16<<<g, 256, 0, stream>>>(mlp_w2, HH, HH, HH, W2T, HH, 0);
        k_precompute<<<2, 256, 0, stream>>>(s2e_w, s2e_b, mlp_w1, mlp_b1, u1, u2, c0);
        k_init<<<(BB * 8 * HH) / 256, 256, 0, stream>>>(embeds0, values0, out, embbf);
    }

    const int gBH = (BB * HH) / 256;   // 16384 blocks for (B,H) elementwise

    for (int step = 0; step < 8; ++step) {
        const int N = 8 + step;

        // summary = mean(embeds[:, :N]) -> bf16
        k_summary<<<gBH, 256, 0, stream>>>(out, sumbf, N, 1.0f / (float)N);

        // q123 = summary @ [qp1|qp2|opq]   (M=8192, N=1536, K=512)
        k_gemm_bf16<<<dim3(BB / TM, (3 * HH) / TN), 256, 0, stream>>>(
            sumbf, HH, WqT, HH, q123, 3 * HH, HH, nullptr, 0);

        // keys = embeds @ key_w + key_b   (M=B*16, N=512, K=512; rows >= N unused)
        k_gemm_bf16<<<dim3((BB * NMAX) / TM, HH / TN), 256, 0, stream>>>(
            embbf, HH, keyWT, HH, keys, HH, HH, key_b, 0);

        // fused attention / ops / mlp-input assembly
        k_attention<<<BB / 8, 256, 0, stream>>>(
            out, keys, q123, operand_ctx, op_ctx, step_emb + (size_t)step * HH,
            qp1_b, qp2_b, opq_b, opsel_w, opsel_b, u1, u2, c0, mlp_w1,
            e12bf, esum, mlpin, N);

        // mlp_in += [e1|e2] @ mlp_w1[0:1024]   (M=8192, N=512, K=1024, accumulate)
        k_gemm_bf16<<<dim3(BB / TM, HH / TN), 256, 0, stream>>>(
            e12bf, 2 * HH, W1abT, 2 * HH, mlpin, HH, 2 * HH, nullptr, 1);

        // exact GELU -> bf16
        k_gelu<<<gBH, 256, 0, stream>>>(mlpin, hidbf);

        // delta = gelu(...) @ mlp_w2 + mlp_b2   (M=8192, N=512, K=512)
        k_gemm_bf16<<<dim3(BB / TM, HH / TN), 256, 0, stream>>>(
            hidbf, HH, W2T, HH, delta, HH, HH, mlp_b2, 0);

        // append new embedding row (d_out + bf16 mirror)
        k_finalize<<<gBH, 256, 0, stream>>>(esum, delta, out, embbf, N);
    }
}
